// FullLSTMModel_29978871726536
// MI455X (gfx1250) — matrix operand order repaired and checked by reading, not compiled, not run on hardware
//
#include <hip/hip_runtime.h>
#include <hip/hip_bf16.h>

typedef __attribute__((ext_vector_type(16))) _Float16 v16h;
typedef __attribute__((ext_vector_type(8)))  _Float16 v8h;
typedef __attribute__((ext_vector_type(8)))  float    v8f;

#define KTOT 800      // 516 (x) + 256 (h) padded to multiple of 32
#define KIN  516
#define KH   256
#define NGATE 1024    // 4*H
#define HDIM 256
#define KB   32       // WMMA K per step
#define TM   64       // batch rows per block
#define NCH  64       // H columns per block (per gate)

// --- CDNA5 async global->LDS copy (ASYNCcnt path), with safe fallback ------
#if __has_builtin(__builtin_amdgcn_global_load_async_to_lds_b128) && \
    __has_builtin(__builtin_amdgcn_s_wait_asynccnt)
// per hipcc diagnostic: params are pointers to 128-bit int vectors,
// src in AS1 (global / __device__), dst in AS3 (LDS)
typedef int b128_t __attribute__((vector_size(16)));
typedef __attribute__((address_space(1))) b128_t as1_b128;
typedef __attribute__((address_space(3))) b128_t as3_b128;
#define ASYNC_B128(gsrc, ldst)                                            \
  __builtin_amdgcn_global_load_async_to_lds_b128(                         \
      (as1_b128*)(uintptr_t)(gsrc), (as3_b128*)(uintptr_t)(ldst), 0, 0)
#define WAIT_ASYNC(n) __builtin_amdgcn_s_wait_asynccnt(n)
#else
// sync fallback: global load -> ds_store (correct under the same barriers)
#define ASYNC_B128(gsrc, ldst) \
  do { *(v8h*)(ldst) = *(const v8h*)(gsrc); } while (0)
#define WAIT_ASYNC(n) ((void)0)
#endif

// ---------------------------------------------------------------------------
// Kernel 0: build f16 weight matrix W[1024][800] = [W_ih | W_hh | 0-pad],
// and fused bias b[1024] = b_ih + b_hh.
// ---------------------------------------------------------------------------
__global__ __launch_bounds__(256) void pack_w_kernel(
    const float* __restrict__ W_ih, const float* __restrict__ W_hh,
    const float* __restrict__ b_ih, const float* __restrict__ b_hh,
    _Float16* __restrict__ Wp, float* __restrict__ bias) {
  int idx = blockIdx.x * 256 + threadIdx.x;
  if (idx < NGATE * KTOT) {
    int n = idx / KTOT;
    int k = idx - n * KTOT;
    float v;
    if (k < KIN)            v = W_ih[n * KIN + k];
    else if (k < KIN + KH)  v = W_hh[n * KH + (k - KIN)];
    else                    v = 0.0f;
    Wp[idx] = (_Float16)v;
  }
  if (idx < NGATE) bias[idx] = b_ih[idx] + b_hh[idx];
}

// ---------------------------------------------------------------------------
// Kernel 1: build f16 activation matrix X[B][800] =
//   [hand_pool(64) | prev_pool(64) | played_pool(4x64) | num_emb(16) |
//    cur_p(16) | prv_p(16) | rt(64) | ht(16) | fm(4) | h(256) | 0-pad(28)]
// 4 batch rows per block; card_emb staged in LDS; masks as 52-bit bitmasks.
// ---------------------------------------------------------------------------
__global__ __launch_bounds__(256) void pack_x_kernel(
    const int* __restrict__ card_list, const int* __restrict__ prev_play,
    const int* __restrict__ played,    const int* __restrict__ player_no,
    const int* __restrict__ prev_player, const int* __restrict__ round_type,
    const int* __restrict__ hand_type, const int* __restrict__ is_first,
    const float* __restrict__ hidden,
    const float* __restrict__ card_emb, const float* __restrict__ nce,
    const float* __restrict__ pemb,     const float* __restrict__ rtemb,
    const float* __restrict__ htemb,    const float* __restrict__ fmemb,
    _Float16* __restrict__ X) {
  __shared__ float cemb[52 * 64];
  __shared__ unsigned long long msk[4][6];

  const int tid = threadIdx.x;
  const int row0 = blockIdx.x * 4;

  for (int i = tid; i < 52 * 64; i += 256) cemb[i] = card_emb[i];

  if (tid < 24) {
    int lr = tid / 6, s = tid % 6;
    int row = row0 + lr;
    const int* p;
    if (s == 0)      p = card_list + (size_t)row * 52;
    else if (s == 1) p = prev_play + (size_t)row * 52;
    else             p = played + (size_t)row * 208 + (s - 2) * 52;
    unsigned long long m = 0ull;
    for (int c = 0; c < 52; ++c)
      if (p[c] != 0) m |= (1ull << c);
    msk[lr][s] = m;
  }
  __syncthreads();

  const int lr = tid >> 6;        // local row 0..3
  const int e  = tid & 63;        // embedding dim 0..63
  const int row = row0 + lr;
  _Float16* Xr = X + (size_t)row * KTOT;

  // masked-max pools (branch-free; empty set -> 0)
  float pool[6];
  for (int s = 0; s < 6; ++s) {
    unsigned long long m = msk[lr][s];
    float v = -1e30f;
    for (int c = 0; c < 52; ++c) {
      float ev = cemb[c * 64 + e];
      v = fmaxf(v, ((m >> c) & 1ull) ? ev : -1e30f);
    }
    pool[s] = (m != 0ull) ? v : 0.0f;
  }
  Xr[e]      = (_Float16)pool[0];
  Xr[64 + e] = (_Float16)pool[1];
  for (int p = 0; p < 4; ++p)
    Xr[128 + p * 64 + e] = (_Float16)pool[2 + p];

  // scalar feature section: x[384..516) -> local j in [0,132)
  const int pn = player_no[row];
  const int pv = prev_player[row];
  const int rt = min(round_type[row], 4);
  const int ht = hand_type[row];
  const int fm = is_first[row];
  int cnt[4];
  for (int p = 0; p < 4; ++p)
    cnt[p] = min(__popcll(msk[lr][2 + p]), 12);

  for (int j = e; j < 132; j += 64) {
    float v;
    if (j < 16) {                       // num_cards embeddings [13,4]
      int p = j >> 2, comp = j & 3;
      v = nce[cnt[p] * 4 + comp];
    } else if (j < 32) {                // current player [4,16]
      v = pemb[pn * 16 + (j - 16)];
    } else if (j < 48) {                // previous player
      v = pemb[pv * 16 + (j - 32)];
    } else if (j < 112) {               // round type [5,64]
      v = rtemb[rt * 64 + (j - 48)];
    } else if (j < 128) {               // hand type [6,16], padding_idx=0
      v = (ht == 0) ? 0.0f : htemb[ht * 16 + (j - 112)];
    } else {                            // first move [2,4]
      v = fmemb[fm * 4 + (j - 128)];
    }
    Xr[384 + j] = (_Float16)v;
  }

  for (int h = e; h < KH; h += 64)
    Xr[KIN + h] = (_Float16)hidden[(size_t)row * KH + h];

  for (int j = 772 + e; j < KTOT; j += 64)
    Xr[j] = (_Float16)0.0f;
}

// ---------------------------------------------------------------------------
// Kernel 2: fused GEMM (WMMA f16->f32) + LSTM cell epilogue.
// Block: 256 threads = 8 wave32. Tile: 64 batch rows x 64 H cols x 4 gates.
// Double-buffered LDS staging fed by GLOBAL_LOAD_ASYNC_TO_LDS_B128
// (ASYNCcnt), overlapping tile t+1 DMA with tile t WMMAs.
// ---------------------------------------------------------------------------
__global__ __launch_bounds__(256) void gemm_lstm_kernel(
    const _Float16* __restrict__ X, const _Float16* __restrict__ W,
    const float* __restrict__ bias, const float* __restrict__ cell,
    float* __restrict__ out, int B) {
  extern __shared__ char smem[];
  // staging (lifetime: K loop)         // Xs: 2 x 64x32 f16   =  8 KB
  _Float16* Xs0 = (_Float16*)smem;      // Ws: 2 x 256x32 f16  = 32 KB
  _Float16* Xs1 = (_Float16*)(smem + 4096);
  _Float16* Ws0 = (_Float16*)(smem + 8192);
  _Float16* Ws1 = (_Float16*)(smem + 8192 + 16384);
  // gate buffer (lifetime: after K loop) 4x64x64 f32 = 64 KB
  float* gbuf = (float*)smem;

  const int tid  = threadIdx.x;
  const int lane = tid & 31;
  const int w    = tid >> 5;
  const int mw   = w & 3;          // M subtile
  const int nh   = w >> 2;         // gate-pair half (0: i,f  1: g,o)
  const int lm   = lane & 15;
  const int half = lane >> 4;      // hi/lo lane half per WMMA layout

  const int m0 = blockIdx.x * TM;  // batch row base
  const int h0 = blockIdx.y * NCH; // H column base

  // per-thread staging coordinates (5 B128 transfers per tile per thread)
  const int xr  = tid >> 2;              // X row 0..63
  const int xkk = (tid & 3) * 8;         // X k-sub 0/8/16/24
  const _Float16* Xg = X + (size_t)(m0 + xr) * KTOT + xkk;

  auto issue_tile = [&](int k0, _Float16* Xsb, _Float16* Wsb) {
    ASYNC_B128(Xg + k0, Xsb + xr * KB + xkk);
    for (int i = 0; i < 4; ++i) {
      int idx = tid + i * 256;
      int c   = idx >> 2;                // 0..255 (gate*64 + local col)
      int kk  = (idx & 3) * 8;
      int gcol = ((c >> 6) << 8) + h0 + (c & 63);
      ASYNC_B128(W + (size_t)gcol * KTOT + k0 + kk, Wsb + c * KB + kk);
    }
  };

  v8f acc[8] = {};

  const int NT = KTOT / KB;              // 25 K-tiles
  issue_tile(0, Xs0, Ws0);               // prologue: tile 0 -> buffer 0

  for (int t = 0; t < NT; ++t) {
    _Float16* Xs = (t & 1) ? Xs1 : Xs0;
    _Float16* Ws = (t & 1) ? Ws1 : Ws0;
    if (t + 1 < NT) {
      // overlap: DMA tile t+1 into the other buffer while computing tile t
      issue_tile((t + 1) * KB, (t & 1) ? Xs0 : Xs1, (t & 1) ? Ws0 : Ws1);
      WAIT_ASYNC(5);     // in-order completion => tile t's 5 copies are done
    } else {
      WAIT_ASYNC(0);
    }
    __syncthreads();     // tile t visible to all waves

    // A fragment: 16x32 f16. lane(lm,half): elems 0..7 = K half*8..,
    // elems 8..15 = K 16+half*8..
    v16h a;
    {
      const _Float16* ap = Xs + (mw * 16 + lm) * KB;
      v8h lo = *(const v8h*)(ap + half * 8);
      v8h hi = *(const v8h*)(ap + 16 + half * 8);
      for (int i = 0; i < 8; ++i) { a[i] = lo[i]; a[8 + i] = hi[i]; }
    }
    // B fragments + WMMA: same K order as A (elems 0..7 = K half*8.., 8..15 = K 16+half*8..)
    for (int gi = 0; gi < 2; ++gi) {
      for (int nt = 0; nt < 4; ++nt) {
        int c = (nh * 2 + gi) * 64 + nt * 16 + lm;
        const _Float16* bp = Ws + c * KB;
        v8h lo = *(const v8h*)(bp + half * 8);
        v8h hi = *(const v8h*)(bp + 16 + half * 8);
        v16h b;
        for (int i = 0; i < 8; ++i) { b[i] = lo[i]; b[8 + i] = hi[i]; }
        acc[gi * 4 + nt] = __builtin_amdgcn_wmma_f32_16x16x32_f16(
            false, a, false, b, (short)0, acc[gi * 4 + nt], false, false);
      }
    }
    __syncthreads();     // all reads done before t+1's DMA lands in this buf
  }

  // dump gate accumulators to LDS: gbuf[gate][row(64)][col(64)]
  for (int gi = 0; gi < 2; ++gi) {
    int gate = nh * 2 + gi;
    for (int nt = 0; nt < 4; ++nt) {
      for (int r = 0; r < 8; ++r) {
        int row = mw * 16 + r + 8 * half;   // C layout: M = r + 8*(lane>=16)
        int col = nt * 16 + lm;
        gbuf[(gate * 64 + row) * 64 + col] = acc[gi * 4 + nt][r];
      }
    }
  }
  __syncthreads();

  // LSTM cell epilogue (torch gate order i,f,g,o)
  for (int idx = tid; idx < TM * NCH; idx += 256) {
    int row  = idx >> 6;
    int col  = idx & 63;
    int grow = m0 + row;
    int hcol = h0 + col;
    float gi_ = gbuf[(0 * 64 + row) * 64 + col] + bias[0 * HDIM + hcol];
    float gf_ = gbuf[(1 * 64 + row) * 64 + col] + bias[1 * HDIM + hcol];
    float gg_ = gbuf[(2 * 64 + row) * 64 + col] + bias[2 * HDIM + hcol];
    float go_ = gbuf[(3 * 64 + row) * 64 + col] + bias[3 * HDIM + hcol];
    float i_ = 1.0f / (1.0f + __expf(-gi_));
    float f_ = 1.0f / (1.0f + __expf(-gf_));
    float g_ = tanhf(gg_);
    float o_ = 1.0f / (1.0f + __expf(-go_));
    float c_old = cell[(size_t)grow * HDIM + hcol];
    float c_new = f_ * c_old + i_ * g_;
    float h_new = o_ * tanhf(c_new);
    out[(size_t)grow * HDIM + hcol] = h_new;                          // h_new
    out[(size_t)B * HDIM + (size_t)grow * HDIM + hcol] = c_new;       // c_new
  }
}

// ---------------------------------------------------------------------------
extern "C" void kernel_launch(void* const* d_in, const int* in_sizes, int n_in,
                              void* d_out, int out_size, void* d_ws, size_t ws_size,
                              hipStream_t stream) {
  const int*   card_list   = (const int*)d_in[0];
  const int*   prev_play   = (const int*)d_in[1];
  const int*   played      = (const int*)d_in[2];
  const int*   player_no   = (const int*)d_in[3];
  const int*   prev_player = (const int*)d_in[4];
  const int*   round_type  = (const int*)d_in[5];
  const int*   hand_type   = (const int*)d_in[6];
  const int*   is_first    = (const int*)d_in[7];
  const float* hidden      = (const float*)d_in[8];
  const float* cell        = (const float*)d_in[9];
  const float* card_emb    = (const float*)d_in[10];
  const float* nce         = (const float*)d_in[11];
  const float* pemb        = (const float*)d_in[12];
  const float* rtemb       = (const float*)d_in[13];
  const float* htemb       = (const float*)d_in[14];
  const float* fmemb       = (const float*)d_in[15];
  const float* W_ih        = (const float*)d_in[16];
  const float* W_hh        = (const float*)d_in[17];
  const float* b_ih        = (const float*)d_in[18];
  const float* b_hh        = (const float*)d_in[19];
  float* out = (float*)d_out;

  const int B = in_sizes[8] / HDIM;   // 16384

  size_t xbytes = (size_t)B * KTOT * sizeof(_Float16);        // 26.2 MB
  size_t wbytes = (size_t)NGATE * KTOT * sizeof(_Float16);    // 1.6 MB
  _Float16* Xp   = (_Float16*)d_ws;
  _Float16* Wp   = (_Float16*)((char*)d_ws + xbytes);
  float*    bias = (float*)((char*)d_ws + xbytes + wbytes);

  {
    int total = NGATE * KTOT;
    pack_w_kernel<<<(total + 255) / 256, 256, 0, stream>>>(
        W_ih, W_hh, b_ih, b_hh, Wp, bias);
  }
  pack_x_kernel<<<B / 4, 256, 0, stream>>>(
      card_list, prev_play, played, player_no, prev_player, round_type,
      hand_type, is_first, hidden, card_emb, nce, pemb, rtemb, htemb, fmemb,
      Xp);
  dim3 grid(B / TM, HDIM / NCH);   // 256 x 4
  gemm_lstm_kernel<<<grid, 256, 65536, stream>>>(Xp, Wp, bias, cell, out, B);
}
